// Attention_26860725469356
// MI455X (gfx1250) — matrix operand order repaired and checked by reading, not compiled, not run on hardware
//
#include <hip/hip_runtime.h>
#include <math.h>

// ---- sizes from the reference ----
#define Bsz 128
#define Rsz 196
#define Csz 2048
#define Hsz 1024
#define Asz 1024
#define BRsz (Bsz * Rsz) // 25088

// Own vector types (avoid HIP vector-class/union quirks)
typedef __attribute__((ext_vector_type(4)))  float        f32x4;
typedef __attribute__((ext_vector_type(4)))  unsigned int u32x4;
typedef __attribute__((ext_vector_type(16))) __bf16       v16bf;
typedef __attribute__((ext_vector_type(8)))  float        v8f;

union Frag16 { v16bf v; u32x4 q[2]; };

__device__ __forceinline__ float gelu_exact_f(float x) {
  return 0.5f * x * (1.0f + erff(x * 0.70710678118654752440f));
}

// CDNA5 async global->LDS copy (no VGPR staging, tracked by ASYNCcnt).
// GV form: global_load_async_to_lds_b128 vdst(LDS byte addr), vaddr(64b), off
__device__ __forceinline__ void async_copy_b128(unsigned lds_off, const void* gaddr) {
  asm volatile("global_load_async_to_lds_b128 %0, %1, off"
               :: "v"(lds_off), "v"(gaddr) : "memory");
}
__device__ __forceinline__ void wait_async0() {
#if __has_builtin(__builtin_amdgcn_s_wait_asynccnt)
  __builtin_amdgcn_s_wait_asynccnt(0);
#else
  asm volatile("s_wait_asynccnt 0" ::: "memory");
#endif
}

// ---------------------------------------------------------------------------
// prep: convert We (A x C, f32) -> bf16 workspace; zero s2 accumulator
// grid = (Asz*Csz/4)/256 = 2048 blocks, 256 threads
// ---------------------------------------------------------------------------
__global__ __launch_bounds__(256) void prep_k(const float* __restrict__ We,
                                              __bf16* __restrict__ WeB,
                                              float* __restrict__ s2) {
  const size_t i = (size_t)blockIdx.x * 256 + threadIdx.x; // float4 index
  const f32x4 f = ((const f32x4*)We)[i];
  __bf16* d = WeB + i * 4;
  d[0] = (__bf16)f.x; d[1] = (__bf16)f.y; d[2] = (__bf16)f.z; d[3] = (__bf16)f.w;
  if (blockIdx.x == 0 && threadIdx.x < Bsz) s2[threadIdx.x] = 0.0f;
}

// ---------------------------------------------------------------------------
// decoder path: s2[b] = sum_a gelu(dh[b,:] . Wd[a,:] + bd[a]) * Wf2[a]
// grid = 4 blocks x 256 threads (one thread per 'a'), 16 b-groups of 8
// ---------------------------------------------------------------------------
__global__ __launch_bounds__(256) void dec_score_k(const float* __restrict__ dh,
                                                   const float* __restrict__ Wd,
                                                   const float* __restrict__ bd,
                                                   const float* __restrict__ Wf,
                                                   float* __restrict__ s2) {
  __shared__ float s2loc[Bsz];
  const int tid = threadIdx.x;
  if (tid < Bsz) s2loc[tid] = 0.0f;
  __syncthreads();

  const int a = blockIdx.x * 256 + tid; // < 1024
  const float bda = bd[a];
  const float wf2 = Wf[Asz + a];

  for (int bg = 0; bg < 16; ++bg) {
    float accv[8];
#pragma unroll
    for (int j = 0; j < 8; ++j) accv[j] = 0.0f;
    for (int h = 0; h < Hsz; h += 4) {
      const f32x4 wd = *(const f32x4*)(Wd + (size_t)a * Hsz + h);
#pragma unroll
      for (int j = 0; j < 8; ++j) {
        const f32x4 dv = *(const f32x4*)(dh + (size_t)(bg * 8 + j) * Hsz + h);
        accv[j] += dv.x * wd.x + dv.y * wd.y + dv.z * wd.z + dv.w * wd.w;
      }
    }
#pragma unroll
    for (int j = 0; j < 8; ++j) {
      const float gl = gelu_exact_f(accv[j] + bda);
      atomicAdd(&s2loc[bg * 8 + j], gl * wf2);
    }
  }
  __syncthreads();
  if (tid < Bsz) atomicAdd(&s2[tid], s2loc[tid]);
}

// ---------------------------------------------------------------------------
// main fused kernel: for 64 rows m of flattened (b,r):
//   scores1[m] = sum_{n<1024} gelu( (image[m,:] . We[n,:]) + be[n] ) * Wf1[n]
// bf16 WMMA 16x16x32, block tile M=64 x N=1024, K-step 32, double-buffered LDS.
// B tile staged via CDNA5 async global->LDS (ASYNCcnt); A tile converted
// f32->bf16 through VGPRs. 512 threads = 16 waves: (mtile 0..3) x (nquad 0..3)
// ---------------------------------------------------------------------------
__global__ __launch_bounds__(512) void gemm_score_k(const float*  __restrict__ image,
                                                    const __bf16* __restrict__ WeB,
                                                    const float*  __restrict__ be,
                                                    const float*  __restrict__ Wf,
                                                    float* __restrict__ scores1) {
  extern __shared__ char smem[];
  __bf16* As = (__bf16*)smem;                              // [2][64*32]
  __bf16* Bs = (__bf16*)(smem + 2 * 64 * 32 * 2);          // [2][1024*32]
  float* blockScores = (float*)(smem + 2 * 64 * 32 * 2 + 2 * 1024 * 32 * 2); // [64]

  const int tid   = threadIdx.x;
  const int wave  = tid >> 5;
  const int lane  = tid & 31;
  const int half  = (lane < 16) ? 0 : 1;
  const int l15   = lane & 15;
  const int mBase = blockIdx.x * 64;

  const int mtile  = wave >> 2; // 0..3
  const int ntquad = wave & 3;  // 0..3

  if (tid < 64) blockScores[tid] = 0.0f;

  // LDS byte address of the B staging area (flat->LDS: low 32 bits)
  const unsigned bsLds = (unsigned)(size_t)(void*)Bs;

  // per-lane fragment source offsets (ISA 7.12.2 16-bit layouts)
  const int arowElem = (mtile * 16 + l15) * 32; // A row in LDS tile
  const int ak0 = half ? 8 : 0;                 // A: K chunks {0..7,16..23} / {8..15,24..31}
  const int bk0 = half ? 8 : 0;                 // B: K chunks {0..7,16..23} / {8..15,24..31}, n = l15

  auto loadA = [&](int buf, int k0) {
    const int e = tid * 4;            // 64*32 = 2048 elems, 4 per thread
    const int row = e >> 5, col = e & 31;
    const f32x4 f = *(const f32x4*)(image + (size_t)(mBase + row) * Csz + k0 + col);
    __bf16* d = As + buf * 2048 + row * 32 + col;
    d[0] = (__bf16)f.x; d[1] = (__bf16)f.y; d[2] = (__bf16)f.z; d[3] = (__bf16)f.w;
  };
  // async B staging: thread copies 2 rows x 64B, 4 x b128 each, straight to LDS
  auto loadB = [&](int buf, int k0) {
#pragma unroll
    for (int i = 0; i < 2; ++i) {
      const int n = tid * 2 + i;
      const char* src = (const char*)(WeB + (size_t)n * Csz + k0);
      const unsigned dst = bsLds + (unsigned)((buf * 32768 + n * 32) * 2);
#pragma unroll
      for (int j = 0; j < 4; ++j) {
        async_copy_b128(dst + j * 16, src + j * 16);
      }
    }
  };

  v8f acc[16];
  const v8f vzero = {0.f, 0.f, 0.f, 0.f, 0.f, 0.f, 0.f, 0.f};
#pragma unroll
  for (int t = 0; t < 16; ++t) acc[t] = vzero;

  loadA(0, 0);
  loadB(0, 0);
  wait_async0();
  __syncthreads();

  const int KSTEPS = Csz / 32; // 64
  for (int ks = 0; ks < KSTEPS; ++ks) {
    const int buf = ks & 1;
    if (ks + 1 < KSTEPS) { loadB(buf ^ 1, (ks + 1) * 32); loadA(buf ^ 1, (ks + 1) * 32); }
    if (ks + 2 < KSTEPS) { // gfx1250 global_prefetch_b8 path
      const int e = tid * 4;
      __builtin_prefetch(image + (size_t)(mBase + (e >> 5)) * Csz + (ks + 2) * 32 + (e & 31), 0, 0);
    }

    Frag16 af;
    const __bf16* ab = As + buf * 2048 + arowElem + ak0;
    af.q[0] = *(const u32x4*)(ab);        // K chunk 0 (8 bf16)
    af.q[1] = *(const u32x4*)(ab + 16);   // K chunk 1 (+16 in K)

#pragma unroll
    for (int t = 0; t < 16; ++t) {
      const int nt = ntquad * 16 + t;
      Frag16 bfm;
      const __bf16* bb = Bs + buf * 32768 + (size_t)(nt * 16 + l15) * 32 + bk0;
      bfm.q[0] = *(const u32x4*)(bb);
      bfm.q[1] = *(const u32x4*)(bb + 16); // K chunk 1 (+16 in K)
      acc[t] = __builtin_amdgcn_wmma_f32_16x16x32_bf16(
          false, af.v, false, bfm.v, (short)0, acc[t], false, false);
    }

    wait_async0();      // our async writes into buf^1 are done before we signal
    __syncthreads();    // everyone done reading buf & writing buf^1
  }

  // epilogue: +be, exact gelu, dot with Wf1, reduce over n
  float wpart[8];
#pragma unroll
  for (int g = 0; g < 8; ++g) wpart[g] = 0.0f;
#pragma unroll
  for (int t = 0; t < 16; ++t) {
    const int n = (ntquad * 16 + t) * 16 + l15;
    const float bev = be[n];
    const float w1 = Wf[n];
#pragma unroll
    for (int g = 0; g < 8; ++g) {
      const float x = acc[t][g] + bev; // row m = mtile*16 + g + half*8
      wpart[g] += gelu_exact_f(x) * w1;
    }
  }
  // butterfly reduce across the 16 lanes of each half (same M set, different n)
#pragma unroll
  for (int m = 1; m < 16; m <<= 1) {
#pragma unroll
    for (int g = 0; g < 8; ++g) wpart[g] += __shfl_xor(wpart[g], m, 32);
  }
  if (l15 == 0) {
    const int mrow = mtile * 16 + half * 8;
#pragma unroll
    for (int g = 0; g < 8; ++g) atomicAdd(&blockScores[mrow + g], wpart[g]);
  }
  __syncthreads();
  if (tid < 64) scores1[mBase + tid] = blockScores[tid];
}

// ---------------------------------------------------------------------------
// softmax over r per batch: alpha[b,r] = softmax(scores1[b,r] + s2[b] + bf)
// grid = 128, 256 threads
// ---------------------------------------------------------------------------
__global__ __launch_bounds__(256) void softmax_k(const float* __restrict__ scores1,
                                                 const float* __restrict__ s2,
                                                 const float* __restrict__ bf,
                                                 float* __restrict__ alpha_out) {
  __shared__ float red[256];
  const int b = blockIdx.x;
  const int tid = threadIdx.x;
  const float base = s2[b] + bf[0];

  float myv = -INFINITY;
  if (tid < Rsz) myv = scores1[b * Rsz + tid] + base;

  red[tid] = myv;
  __syncthreads();
  for (int s = 128; s > 0; s >>= 1) {
    if (tid < s) red[tid] = fmaxf(red[tid], red[tid + s]);
    __syncthreads();
  }
  const float mx = red[0];
  __syncthreads();

  float e = 0.0f;
  if (tid < Rsz) e = expf(myv - mx);
  red[tid] = e;
  __syncthreads();
  for (int s = 128; s > 0; s >>= 1) {
    if (tid < s) red[tid] += red[tid + s];
    __syncthreads();
  }
  const float inv = 1.0f / red[0];
  if (tid < Rsz) alpha_out[b * Rsz + tid] = e * inv;
}

// ---------------------------------------------------------------------------
// attended_image[b,c] = sum_r image[b,r,c] * alpha[b,r]
// grid = (2, 128), 256 threads, float4 per thread (1024 c per block)
// ---------------------------------------------------------------------------
__global__ __launch_bounds__(256) void attend_k(const float* __restrict__ image,
                                                const float* __restrict__ alpha,
                                                float* __restrict__ out) {
  __shared__ float al[Rsz];
  const int b = blockIdx.y;
  if (threadIdx.x < Rsz) al[threadIdx.x] = alpha[b * Rsz + threadIdx.x];
  __syncthreads();

  const int c = blockIdx.x * 1024 + threadIdx.x * 4;
  f32x4 acc = {0.f, 0.f, 0.f, 0.f};
  for (int r = 0; r < Rsz; ++r) {
    const f32x4 im = *(const f32x4*)(image + ((size_t)b * Rsz + r) * Csz + c);
    const float a = al[r];
    acc.x += im.x * a; acc.y += im.y * a; acc.z += im.z * a; acc.w += im.w * a;
  }
  *(f32x4*)(out + (size_t)b * Csz + c) = acc;
}

// ---------------------------------------------------------------------------
extern "C" void kernel_launch(void* const* d_in, const int* in_sizes, int n_in,
                              void* d_out, int out_size, void* d_ws, size_t ws_size,
                              hipStream_t stream) {
  const float* image = (const float*)d_in[0]; // (B,R,C)
  const float* dh    = (const float*)d_in[1]; // (B,H)
  const float* We    = (const float*)d_in[2]; // (A,C)
  const float* be    = (const float*)d_in[3]; // (A,)
  const float* Wd    = (const float*)d_in[4]; // (A,H)
  const float* bd    = (const float*)d_in[5]; // (A,)
  const float* Wf    = (const float*)d_in[6]; // (1,2A)
  const float* bf    = (const float*)d_in[7]; // (1,)
  float* out = (float*)d_out; // [attended (B*C)] ++ [alpha (B*R)]

  char* ws = (char*)d_ws;
  __bf16* WeB    = (__bf16*)ws;                                  // 4 MB bf16 We
  float*  scores1 = (float*)(ws + (size_t)Asz * Csz * 2);        // 25088 f32
  float*  s2      = scores1 + BRsz;                              // 128 f32

  // 1) We -> bf16, zero s2
  prep_k<<<(Asz * Csz / 4) / 256, 256, 0, stream>>>(We, WeB, s2);
  // 2) decoder-side scalar scores s2[b]
  dec_score_k<<<Asz / 256, 256, 0, stream>>>(dh, Wd, bd, Wf, s2);
  // 3) fused bf16-WMMA GEMM + gelu + Wf1 reduction -> scores1
  const size_t smem = (size_t)2 * 64 * 32 * 2 + (size_t)2 * 1024 * 32 * 2 + 64 * 4;
  gemm_score_k<<<BRsz / 64, 512, smem, stream>>>(image, WeB, be, Wf, scores1);
  // 4) softmax -> alpha (written straight into d_out tail)
  softmax_k<<<Bsz, 256, 0, stream>>>(scores1, s2, bf, out + (size_t)Bsz * Csz);
  // 5) attended image
  attend_k<<<dim3(2, Bsz), 256, 0, stream>>>(image, out + (size_t)Bsz * Csz, out);
}
